// EGNNLayer_3539053052443
// MI455X (gfx1250) — matrix operand order repaired and checked by reading, not compiled, not run on hardware
//
#include <hip/hip_runtime.h>
#include <hip/hip_bf16.h>
#include <math.h>

// ---------------------------------------------------------------------------
// EGNN layer for MI455X (gfx1250): f16 WMMA (16x16x32) matmuls, fp32 accum.
// ---------------------------------------------------------------------------

typedef _Float16 half_t;
typedef __attribute__((ext_vector_type(16))) _Float16 v16h;
typedef __attribute__((ext_vector_type(8)))  _Float16 v8h;
typedef __attribute__((ext_vector_type(4)))  _Float16 v4h;
typedef __attribute__((ext_vector_type(8)))  float    v8f;

// Intra-wave LDS ordering: per-wave private LDS regions, so a dscnt wait +
// compiler memory fence is sufficient (no block barrier -> safe in
// grid-stride loops where waves diverge).
#define WAVE_SYNC() asm volatile("s_wait_dscnt 0" ::: "memory")

__device__ __forceinline__ float silu_f(float x) {
  return x / (1.0f + __expf(-x));
}

__device__ __forceinline__ v8f wmma16(v16h a, v16h b, v8f c) {
  // D = A(16x32 f16) * B(32x16 f16) + C(16x16 f32)
  return __builtin_amdgcn_wmma_f32_16x16x32_f16(false, a, false, b,
                                                (short)0, c, false, false);
}

// A-fragment from a row-major 16x128 f16 tile in LDS (ISA 16-bit A layout):
// lane m=lane&15, kb=(lane>>4)*8; halves 0..7 -> K=kb..kb+7,
// halves 8..15 -> K=kb+16..kb+23  => two contiguous 16B LDS reads.
__device__ __forceinline__ v16h load_a_frag(const half_t* tile, int chunk, int lane) {
  const int m  = lane & 15;
  const int kb = (lane >> 4) * 8;
  const half_t* p = tile + m * 128 + chunk * 32 + kb;
  union { v16h v; v8h h[2]; } u;
  u.h[0] = *(const v8h*)(p);
  u.h[1] = *(const v8h*)(p + 16);
  return u.v;
}

// B-fragments pre-shuffled in workspace: frag-major, 32 lanes x 16 halves.
__device__ __forceinline__ v16h load_b_frag(const v16h* base, int frag, int lane) {
  return base[(size_t)frag * 32 + lane];
}

__device__ __forceinline__ void atomic_add_f32(float* p, float v) {
  __hip_atomic_fetch_add(p, v, __ATOMIC_RELAXED, __HIP_MEMORY_SCOPE_AGENT);
}

// ---------------------------------------------------------------------------
// Kernel 0: zero agg buffer, copy coords -> output coords section.
// ---------------------------------------------------------------------------
__global__ void egnn_init_kernel(const float* __restrict__ coords,
                                 float* __restrict__ agg,
                                 float* __restrict__ out_coords,
                                 int nAgg, int nCoord) {
  int i = blockIdx.x * blockDim.x + threadIdx.x;
  int stride = gridDim.x * blockDim.x;
  for (int x = i; x < nAgg; x += stride) agg[x] = 0.0f;
  for (int x = i; x < nCoord; x += stride) out_coords[x] = coords[x];
}

// ---------------------------------------------------------------------------
// Kernel 1: pack fp32 weight matrix [K x ncols] into f16 B-fragment layout.
// frag = nt*nchunks + chunk; per lane n=lane&15, kb=(lane>>4)*8,
// half i -> element W[chunk*32 + kb + (i&7) + (i>>3)*16][nt*16 + n].
// ---------------------------------------------------------------------------
__global__ void egnn_pack_kernel(const float* __restrict__ W,
                                 half_t* __restrict__ dst,
                                 int nchunks, int ncols, int total) {
  for (int x = blockIdx.x * blockDim.x + threadIdx.x; x < total;
       x += gridDim.x * blockDim.x) {
    int i    = x & 15;
    int lane = (x >> 4) & 31;
    int frag = x >> 9;
    int chunk = frag % nchunks;
    int nt    = frag / nchunks;
    int k   = chunk * 32 + ((lane >> 4) * 8) + (i & 7) + ((i >> 3) * 16);
    int col = nt * 16 + (lane & 15);
    dst[x] = (half_t)W[(size_t)k * ncols + col];
  }
}

// ---------------------------------------------------------------------------
// Kernel 2: edge pipeline. One wave = 16 edges.
//   hidden = silu(h[row]@W1a + h[col]@W1b + radial*w1r + mb1)   (WMMA)
//   msg    = hidden@mw2 + mb2                                    (WMMA)
//   agg[row] += msg (atomics);  gate = silu(msg@cw1+cb1)@cw2+cb2 (WMMA+shfl)
//   coords_out[row] += gate * diff / (radial+1e-8) (atomics)
// ---------------------------------------------------------------------------
__global__ __launch_bounds__(256)
void egnn_edge_kernel(const float* __restrict__ h,
                      const float* __restrict__ coords,
                      const int*   __restrict__ edges, int E,
                      const v16h* __restrict__ fw1a, const v16h* __restrict__ fw1b,
                      const v16h* __restrict__ fw2,  const v16h* __restrict__ fcw1,
                      const float* __restrict__ w1r, const float* __restrict__ mb1,
                      const float* __restrict__ mb2, const float* __restrict__ cw2,
                      const float* __restrict__ cb1, const float* __restrict__ cb2,
                      float* __restrict__ agg, float* __restrict__ out_coords,
                      int numTiles) {
  __shared__ half_t s_tile[8][16 * 128];   // 32 KB, per-wave 16x128 f16 tile
  __shared__ int    s_row[8][16];
  __shared__ int    s_col[8][16];
  __shared__ float  s_rad[8][16];
  __shared__ float  s_cw[8][16];

  const int lane = threadIdx.x & 31;
  const int wave = threadIdx.x >> 5;
  const int n    = lane & 15;
  const int hi   = lane >> 4;
  half_t* tile = s_tile[wave];

  for (int t = blockIdx.x * 8 + wave; t < numTiles; t += gridDim.x * 8) {
    const int base = t * 16;

    // --- edge geometry (lanes 0..15, one edge each) ---
    float dx = 0.f, dy = 0.f, dz = 0.f, rad = 0.f;
    if (lane < 16) {
      int e = base + lane; if (e >= E) e = E - 1;
      int r = edges[e], c = edges[E + e];
      dx = coords[r * 3 + 0] - coords[c * 3 + 0];
      dy = coords[r * 3 + 1] - coords[c * 3 + 1];
      dz = coords[r * 3 + 2] - coords[c * 3 + 2];
      rad = sqrtf(dx * dx + dy * dy + dz * dz);
      s_row[wave][lane] = r; s_col[wave][lane] = c; s_rad[wave][lane] = rad;
    }
    WAVE_SYNC();

    // --- stage h[row] tile (f32 -> f16), build A_row fragments ---
    for (int m = 0; m < 16; ++m) {
      int r = s_row[wave][m];
      float4 v = *(const float4*)(h + (size_t)r * 128 + lane * 4);
      v4h hv = { (half_t)v.x, (half_t)v.y, (half_t)v.z, (half_t)v.w };
      *(v4h*)(tile + m * 128 + lane * 4) = hv;
    }
    WAVE_SYNC();
    v16h aR[4];
#pragma unroll
    for (int c2 = 0; c2 < 4; ++c2) aR[c2] = load_a_frag(tile, c2, lane);
    WAVE_SYNC();

    // --- stage h[col] tile, build A_col fragments ---
    for (int m = 0; m < 16; ++m) {
      int c = s_col[wave][m];
      float4 v = *(const float4*)(h + (size_t)c * 128 + lane * 4);
      v4h hv = { (half_t)v.x, (half_t)v.y, (half_t)v.z, (half_t)v.w };
      *(v4h*)(tile + m * 128 + lane * 4) = hv;
    }
    WAVE_SYNC();
    v16h aC[4];
#pragma unroll
    for (int c2 = 0; c2 < 4; ++c2) aC[c2] = load_a_frag(tile, c2, lane);
    WAVE_SYNC();

    // --- layer 1: hidden = silu(...) -> tile (f16, C->A relayout via LDS) ---
#pragma unroll
    for (int nt = 0; nt < 8; ++nt) {
      v8f acc = {0, 0, 0, 0, 0, 0, 0, 0};
#pragma unroll
      for (int c2 = 0; c2 < 4; ++c2)
        acc = wmma16(aR[c2], load_b_frag(fw1a, nt * 4 + c2, lane), acc);
#pragma unroll
      for (int c2 = 0; c2 < 4; ++c2)
        acc = wmma16(aC[c2], load_b_frag(fw1b, nt * 4 + c2, lane), acc);
      float bn = mb1[nt * 16 + n];
      float wr = w1r[nt * 16 + n];
#pragma unroll
      for (int r = 0; r < 8; ++r) {
        float x = acc[r] + bn + s_rad[wave][r + 8 * hi] * wr;
        tile[(r + 8 * hi) * 128 + nt * 16 + n] = (half_t)silu_f(x);
      }
    }
    WAVE_SYNC();
    v16h aH[4];
#pragma unroll
    for (int c2 = 0; c2 < 4; ++c2) aH[c2] = load_a_frag(tile, c2, lane);
    WAVE_SYNC();

    // --- layer 2: messages; scatter to agg; keep f16 copy in tile ---
#pragma unroll
    for (int nt = 0; nt < 8; ++nt) {
      v8f acc = {0, 0, 0, 0, 0, 0, 0, 0};
#pragma unroll
      for (int c2 = 0; c2 < 4; ++c2)
        acc = wmma16(aH[c2], load_b_frag(fw2, nt * 4 + c2, lane), acc);
      float bn = mb2[nt * 16 + n];
#pragma unroll
      for (int r = 0; r < 8; ++r) {
        float v = acc[r] + bn;
        int m = r + 8 * hi;
        if (base + m < E)
          atomic_add_f32(&agg[(size_t)s_row[wave][m] * 128 + nt * 16 + n], v);
        tile[m * 128 + nt * 16 + n] = (half_t)v;
      }
    }
    WAVE_SYNC();
    v16h aM[4];
#pragma unroll
    for (int c2 = 0; c2 < 4; ++c2) aM[c2] = load_a_frag(tile, c2, lane);

    // --- coord gate: silu(msg@cw1+cb1) . cw2 ---
    float cwa[8] = {0, 0, 0, 0, 0, 0, 0, 0};
#pragma unroll
    for (int nt = 0; nt < 4; ++nt) {
      v8f acc = {0, 0, 0, 0, 0, 0, 0, 0};
#pragma unroll
      for (int c2 = 0; c2 < 4; ++c2)
        acc = wmma16(aM[c2], load_b_frag(fcw1, nt * 4 + c2, lane), acc);
      float bn = cb1[nt * 16 + n];
      float w2 = cw2[nt * 16 + n];
#pragma unroll
      for (int r = 0; r < 8; ++r) cwa[r] += silu_f(acc[r] + bn) * w2;
    }
#pragma unroll
    for (int r = 0; r < 8; ++r) {
      float s = cwa[r];
      s += __shfl_xor(s, 1, 32);
      s += __shfl_xor(s, 2, 32);
      s += __shfl_xor(s, 4, 32);
      s += __shfl_xor(s, 8, 32);
      if (n == 0) s_cw[wave][r + 8 * hi] = s + cb2[0];
    }
    WAVE_SYNC();

    // --- coordinate update (lanes 0..15) ---
    if (lane < 16 && base + lane < E) {
      float w = s_cw[wave][lane];
      float inv = 1.0f / (rad + 1e-8f);
      int r = s_row[wave][lane];
      atomic_add_f32(&out_coords[r * 3 + 0], w * dx * inv);
      atomic_add_f32(&out_coords[r * 3 + 1], w * dy * inv);
      atomic_add_f32(&out_coords[r * 3 + 2], w * dz * inv);
    }
    WAVE_SYNC();
  }
}

// ---------------------------------------------------------------------------
// Kernel 3: node update: h_new = silu([h | agg] @ nw1 + nb1) @ nw2 + nb2.
// One wave = 16 nodes (contiguous).
// ---------------------------------------------------------------------------
__global__ __launch_bounds__(256)
void egnn_node_kernel(const float* __restrict__ h,
                      const float* __restrict__ agg,
                      const v16h* __restrict__ fnw1, const v16h* __restrict__ fnw2,
                      const float* __restrict__ nb1, const float* __restrict__ nb2,
                      float* __restrict__ out_h, int N, int numTiles) {
  __shared__ half_t s_tile[8][16 * 128];

  const int lane = threadIdx.x & 31;
  const int wave = threadIdx.x >> 5;
  const int n    = lane & 15;
  const int hi   = lane >> 4;
  half_t* tile = s_tile[wave];

  for (int t = blockIdx.x * 8 + wave; t < numTiles; t += gridDim.x * 8) {
    const int base = t * 16;

    // stage h tile
    for (int m = 0; m < 16; ++m) {
      int node = base + m; if (node >= N) node = N - 1;
      float4 v = *(const float4*)(h + (size_t)node * 128 + lane * 4);
      v4h hv = { (half_t)v.x, (half_t)v.y, (half_t)v.z, (half_t)v.w };
      *(v4h*)(tile + m * 128 + lane * 4) = hv;
    }
    WAVE_SYNC();
    v16h aH[4];
#pragma unroll
    for (int c2 = 0; c2 < 4; ++c2) aH[c2] = load_a_frag(tile, c2, lane);
    WAVE_SYNC();

    // stage agg tile
    for (int m = 0; m < 16; ++m) {
      int node = base + m; if (node >= N) node = N - 1;
      float4 v = *(const float4*)(agg + (size_t)node * 128 + lane * 4);
      v4h hv = { (half_t)v.x, (half_t)v.y, (half_t)v.z, (half_t)v.w };
      *(v4h*)(tile + m * 128 + lane * 4) = hv;
    }
    WAVE_SYNC();
    v16h aA[4];
#pragma unroll
    for (int c2 = 0; c2 < 4; ++c2) aA[c2] = load_a_frag(tile, c2, lane);
    WAVE_SYNC();

    // layer 1 (K = 256: 4 chunks from h, 4 from agg) + silu -> tile
#pragma unroll
    for (int nt = 0; nt < 8; ++nt) {
      v8f acc = {0, 0, 0, 0, 0, 0, 0, 0};
#pragma unroll
      for (int c2 = 0; c2 < 4; ++c2)
        acc = wmma16(aH[c2], load_b_frag(fnw1, nt * 8 + c2, lane), acc);
#pragma unroll
      for (int c2 = 0; c2 < 4; ++c2)
        acc = wmma16(aA[c2], load_b_frag(fnw1, nt * 8 + 4 + c2, lane), acc);
      float bn = nb1[nt * 16 + n];
#pragma unroll
      for (int r = 0; r < 8; ++r)
        tile[(r + 8 * hi) * 128 + nt * 16 + n] = (half_t)silu_f(acc[r] + bn);
    }
    WAVE_SYNC();
    v16h aS[4];
#pragma unroll
    for (int c2 = 0; c2 < 4; ++c2) aS[c2] = load_a_frag(tile, c2, lane);

    // layer 2 -> out
#pragma unroll
    for (int nt = 0; nt < 8; ++nt) {
      v8f acc = {0, 0, 0, 0, 0, 0, 0, 0};
#pragma unroll
      for (int c2 = 0; c2 < 4; ++c2)
        acc = wmma16(aS[c2], load_b_frag(fnw2, nt * 4 + c2, lane), acc);
      float bn = nb2[nt * 16 + n];
#pragma unroll
      for (int r = 0; r < 8; ++r) {
        int node = base + r + 8 * hi;
        if (node < N) out_h[(size_t)node * 128 + nt * 16 + n] = acc[r] + bn;
      }
    }
    WAVE_SYNC();
  }
}

// ---------------------------------------------------------------------------
// Host launch
// ---------------------------------------------------------------------------
extern "C" void kernel_launch(void* const* d_in, const int* in_sizes, int n_in,
                              void* d_out, int out_size, void* d_ws, size_t ws_size,
                              hipStream_t stream) {
  const float* h      = (const float*)d_in[0];
  const float* coords = (const float*)d_in[1];
  const int*   edges  = (const int*)d_in[2];
  const float* mw1 = (const float*)d_in[3];
  const float* mb1 = (const float*)d_in[4];
  const float* mw2 = (const float*)d_in[5];
  const float* mb2 = (const float*)d_in[6];
  const float* nw1 = (const float*)d_in[7];
  const float* nb1 = (const float*)d_in[8];
  const float* nw2 = (const float*)d_in[9];
  const float* nb2 = (const float*)d_in[10];
  const float* cw1 = (const float*)d_in[11];
  const float* cb1 = (const float*)d_in[12];
  const float* cw2 = (const float*)d_in[13];
  const float* cb2 = (const float*)d_in[14];

  const int N = in_sizes[0] / 128;
  const int E = in_sizes[2] / 2;

  float* out_h      = (float*)d_out;
  float* out_coords = out_h + (size_t)N * 128;

  // workspace layout: [agg NxD f32][f16 B-fragment weight buffers]
  float*  agg = (float*)d_ws;
  half_t* fp  = (half_t*)((char*)d_ws + (size_t)N * 128 * sizeof(float));
  half_t* fw1a = fp; fp += 4 * 8 * 512;  // mw1 rows 0..127
  half_t* fw1b = fp; fp += 4 * 8 * 512;  // mw1 rows 128..255
  half_t* fw2  = fp; fp += 4 * 8 * 512;  // mw2
  half_t* fcw1 = fp; fp += 4 * 4 * 512;  // cw1 (128x64)
  half_t* fnw1 = fp; fp += 8 * 8 * 512;  // nw1 (256x128)
  half_t* fnw2 = fp; fp += 4 * 8 * 512;  // nw2

  // init: zero agg, copy coords into output
  {
    int nAgg = N * 128, nCoord = N * 3;
    int blocks = (nAgg + 255) / 256;
    egnn_init_kernel<<<blocks, 256, 0, stream>>>(coords, agg, out_coords, nAgg, nCoord);
  }

  // pack weights into f16 fragment layout
  {
    struct { const float* W; half_t* dst; int nc, nt, ncols; } jobs[6] = {
      { mw1,             fw1a, 4, 8, 128 },
      { mw1 + 128 * 128, fw1b, 4, 8, 128 },
      { mw2,             fw2,  4, 8, 128 },
      { cw1,             fcw1, 4, 4, 64  },
      { nw1,             fnw1, 8, 8, 128 },
      { nw2,             fnw2, 4, 8, 128 },
    };
    for (int j = 0; j < 6; ++j) {
      int total = jobs[j].nc * jobs[j].nt * 512;
      egnn_pack_kernel<<<(total + 255) / 256, 256, 0, stream>>>(
          jobs[j].W, jobs[j].dst, jobs[j].nc, jobs[j].ncols, total);
    }
  }

  // edge pipeline
  {
    int eTiles = (E + 15) / 16;
    int blocks = (eTiles + 7) / 8;
    egnn_edge_kernel<<<blocks, 256, 0, stream>>>(
        h, coords, edges, E,
        (const v16h*)fw1a, (const v16h*)fw1b, (const v16h*)fw2, (const v16h*)fcw1,
        mw1 + (size_t)256 * 128, mb1, mb2, cw2, cb1, cb2,
        agg, out_coords, eTiles);
  }

  // node update
  {
    int nTiles = (N + 15) / 16;
    int blocks = (nTiles + 7) / 8;
    egnn_node_kernel<<<blocks, 256, 0, stream>>>(
        h, agg, (const v16h*)fnw1, (const v16h*)fnw2, nb1, nb2,
        out_h, N, nTiles);
  }

  (void)n_in; (void)out_size; (void)ws_size;
}
